// E87SparseBlockCellFast_11416023073341
// MI455X (gfx1250) — matrix operand order repaired and checked by reading, not compiled, not run on hardware
//
#include <hip/hip_runtime.h>
#include <hip/hip_bf16.h>
#include <math.h>

typedef __attribute__((ext_vector_type(16))) __bf16        v16bf;
typedef __attribute__((ext_vector_type(8)))  float         v8f;
typedef __attribute__((ext_vector_type(8)))  unsigned int  v8u;

#define DIM     1024
#define NSTATE  32
#define NBLOCKS 8
#define TT      1024
#define BB      64
#define NROWS   (TT * BB)
#define NSEG    51          // 1 router + 32 kv + 16 beta + 2 q  (16-wide N tiles)
#define RS      1032        // LDS A-tile row stride in bf16 elements (pad kills bank conflicts)

static __device__ __forceinline__ unsigned pkbf16(float a, float b) {
  // lowers to v_cvt_pk_bf16_f32
  unsigned short ua = __builtin_bit_cast(unsigned short, (__bf16)a);
  unsigned short ub = __builtin_bit_cast(unsigned short, (__bf16)b);
  return (unsigned)ua | ((unsigned)ub << 16);
}

static __device__ __forceinline__ void seg_info(int seg, int& colBase, int& segRows, int& kind) {
  if (seg == 0)       { colBase = 0;             segRows = 8;   kind = 0; }
  else if (seg <= 32) { colBase = (seg - 1)*16;  segRows = 512; kind = 1; }
  else if (seg <= 48) { colBase = (seg - 33)*16; segRows = 256; kind = 2; }
  else                { colBase = (seg - 49)*16; segRows = 32;  kind = 3; }
}

static __device__ __forceinline__ v16bf pack2(uint4 t0, uint4 t1) {
  v8u u;
  u[0]=t0.x; u[1]=t0.y; u[2]=t0.z; u[3]=t0.w;
  u[4]=t1.x; u[5]=t1.y; u[6]=t1.z; u[7]=t1.w;
  return __builtin_bit_cast(v16bf, u);
}

// ---------------------------------------------------------------------------
// Kernel 0: pre-swizzle concatenated W into bf16 in exact WMMA B-fragment
// lane order: bw[((seg*32 + ks)*32 + lane)*16 + e], lane=(hi,lo) -> column
// c = colBase+lo, K = ks*32 + hi*16 + e  (contiguous k order, same as the A fragments).
// One-time 1.6 MB; L2-resident for the whole GEMM.
// ---------------------------------------------------------------------------
__global__ __launch_bounds__(256)
void wswizzle_kernel(const float* __restrict__ Wr, const float* __restrict__ Wkv,
                     const float* __restrict__ Wb, const float* __restrict__ Wq,
                     __bf16* __restrict__ bw) {
  const int g    = blockIdx.x * 256 + threadIdx.x;   // 51*32*32 threads exactly
  const int lane = g & 31;
  const int ks   = (g >> 5) & 31;
  const int seg  = g >> 10;
  const int lo   = lane & 15;
  const int hi   = lane >> 4;

  int colBase, segRows, kind;
  seg_info(seg, colBase, segRows, kind);
  const float* W = (kind == 0) ? Wr : (kind == 1) ? Wkv : (kind == 2) ? Wb : Wq;

  const int  c  = colBase + lo;
  const bool cv = (c < segRows);

  float4 b0 = {0,0,0,0}, b1 = {0,0,0,0}, b2 = {0,0,0,0}, b3 = {0,0,0,0};
  if (cv) {
    const float4* p = (const float4*)(W + (size_t)c * DIM + ks * 32 + hi * 16);
    b0 = p[0]; b1 = p[1]; b2 = p[2]; b3 = p[3];
  }
  uint4 s0 = { pkbf16(b0.x,b0.y), pkbf16(b0.z,b0.w), pkbf16(b1.x,b1.y), pkbf16(b1.z,b1.w) };
  uint4 s1 = { pkbf16(b2.x,b2.y), pkbf16(b2.z,b2.w), pkbf16(b3.x,b3.y), pkbf16(b3.z,b3.w) };
  uint4* dst = (uint4*)(bw + ((size_t)(seg * 32 + ks) * 32 + lane) * 16);
  dst[0] = s0;
  dst[1] = s1;
}

// ---------------------------------------------------------------------------
// Kernel 1: projection GEMM. One WG (8 waves) owns a 64-row M block:
//   phase 1: stage x tile (64x1024) fp32 -> bf16 into LDS (x read ONCE)
//   phase 2: wave w computes N tiles {w, w+8, ...}; per K-step: prefetched
//            B (2x b128 global), 4 A frags loaded into DISTINCT regs
//            (clustered ds_load, single dscnt wait), then 4 back-to-back
//            v_wmma_f32_16x16x32_bf16.
// ---------------------------------------------------------------------------
__global__ __launch_bounds__(256, 4)
void proj_wmma_kernel(const float* __restrict__ x,
                      const __bf16* __restrict__ bw,
                      const float* __restrict__ bbeta,
                      float* __restrict__ logitsb,
                      float* __restrict__ kvb,
                      float* __restrict__ betab,
                      float* __restrict__ qb) {
  __shared__ __align__(16) unsigned short Abuf[64 * RS];   // ~129 KB of 320 KB/WGP

  const int tid = threadIdx.x;
  const int m0  = blockIdx.x * 64;

  // ---- phase 1: cooperative stage + convert ----
#pragma unroll 4
  for (int c = tid; c < 64 * 128; c += 256) {
    const int row = c >> 7;
    const int col = (c & 127) * 8;
    const float4* p = (const float4*)(x + (size_t)(m0 + row) * DIM + col);
    const float4 a0 = p[0], a1 = p[1];
    uint4 s = { pkbf16(a0.x,a0.y), pkbf16(a0.z,a0.w), pkbf16(a1.x,a1.y), pkbf16(a1.z,a1.w) };
    *(uint4*)(&Abuf[row * RS + col]) = s;
  }
  __syncthreads();

  const int wave = tid >> 5;
  const int lane = tid & 31;
  const int lo   = lane & 15;
  const int hi   = lane >> 4;

  for (int seg = wave; seg < NSEG; seg += 8) {
    int colBase, segRows, kind;
    seg_info(seg, colBase, segRows, kind);
    const int  c  = colBase + lo;
    const bool cv = (c < segRows);

    const __bf16* bseg = bw + ((size_t)seg * 32 * 32 + lane) * 16;  // +ks*512 elems per step

    v8f acc0 = {}, acc1 = {}, acc2 = {}, acc3 = {};

    // prefetch B fragment for ks=0
    v16bf bf;
    {
      const uint4* bp = (const uint4*)bseg;
      bf = pack2(bp[0], bp[1]);
    }

    for (int ks = 0; ks < 32; ks++) {
      const int k0 = ks * 32;
      const int ab = k0 + hi * 16;

      // all 4 A fragments -> distinct registers (clustered ds_load_b128 x8)
      const unsigned short* a0p = &Abuf[(0 * 16 + lo) * RS + ab];
      const unsigned short* a1p = &Abuf[(1 * 16 + lo) * RS + ab];
      const unsigned short* a2p = &Abuf[(2 * 16 + lo) * RS + ab];
      const unsigned short* a3p = &Abuf[(3 * 16 + lo) * RS + ab];
      const v16bf af0 = pack2(*(const uint4*)a0p, *(const uint4*)(a0p + 8));
      const v16bf af1 = pack2(*(const uint4*)a1p, *(const uint4*)(a1p + 8));
      const v16bf af2 = pack2(*(const uint4*)a2p, *(const uint4*)(a2p + 8));
      const v16bf af3 = pack2(*(const uint4*)a3p, *(const uint4*)(a3p + 8));

      // prefetch next B fragment (wrap at the end: branch-free, in-bounds)
      v16bf bfn;
      {
        const uint4* bp = (const uint4*)(bseg + ((ks + 1) & 31) * 512);
        bfn = pack2(bp[0], bp[1]);
      }

      acc0 = __builtin_amdgcn_wmma_f32_16x16x32_bf16(false, af0, false, bf, (short)0, acc0, false, false);
      acc1 = __builtin_amdgcn_wmma_f32_16x16x32_bf16(false, af1, false, bf, (short)0, acc1, false, false);
      acc2 = __builtin_amdgcn_wmma_f32_16x16x32_bf16(false, af2, false, bf, (short)0, acc2, false, false);
      acc3 = __builtin_amdgcn_wmma_f32_16x16x32_bf16(false, af3, false, bf, (short)0, acc3, false, false);

      bf = bfn;
    }

    // ---- epilogue: C/D layout => VGPR r holds M = r + 8*hi, N = lo ----
    if (cv) {
      const v8f accs[4] = { acc0, acc1, acc2, acc3 };
#pragma unroll
      for (int tt = 0; tt < 4; tt++) {
#pragma unroll
        for (int r = 0; r < 8; r++) {
          const int row = m0 + tt * 16 + hi * 8 + r;
          float v = accs[tt][r];
          if (kind == 0) {
            logitsb[(size_t)row * 8 + c] = v;             // ROUTER_TEMP == 1.0
          } else if (kind == 1) {
            kvb[(size_t)row * 512 + c] = v;
          } else if (kind == 2) {
            float s = v + bbeta[c];
            betab[(size_t)row * 256 + c] = 1.0f / (1.0f + __expf(-s));
          } else {
            qb[(size_t)row * 32 + c] = v;
          }
        }
      }
    }
  }
}

// ---------------------------------------------------------------------------
// Kernel 2: sequential scan. One WG per batch element (64 WGs), 8 waves = 8
// state blocks, lane = state row i, S block in 32 VGPRs/lane.
// ---------------------------------------------------------------------------
__global__ __launch_bounds__(256)
void scan_kernel(const float* __restrict__ S0,
                 const float* __restrict__ logitsb,
                 const float* __restrict__ kvb,
                 const float* __restrict__ betab,
                 const float* __restrict__ qb,
                 float* __restrict__ out) {  // [outputs (T,B,32)] ++ [S_final (B,8,32,32)]
  const int b    = blockIdx.x;
  const int n    = threadIdx.x >> 5;
  const int lane = threadIdx.x & 31;

  __shared__ float lds_bo[NBLOCKS * NSTATE];

  float S[32];
  {
    const float* sp = S0 + (((size_t)b * NBLOCKS + n) * NSTATE + lane) * NSTATE;
#pragma unroll
    for (int j = 0; j < 32; j++) S[j] = sp[j];
  }

  for (int t = 0; t < TT; t++) {
    const size_t row = (size_t)t * BB + b;

    const float kb = kvb[row * 512 + n * 64 + lane];
    const float vb = kvb[row * 512 + n * 64 + 32 + lane];
    const float be = betab[row * 256 + n * 32 + lane];
    const float qv = qb[row * 32 + lane];
    const float lg = logitsb[row * 8 + (lane & 7)];

    float l[8], e[8];
#pragma unroll
    for (int i = 0; i < 8; i++) l[i] = __shfl(lg, i);
    float mx = l[0];
#pragma unroll
    for (int i = 1; i < 8; i++) mx = fmaxf(mx, l[i]);
    float esum = 0.0f;
#pragma unroll
    for (int i = 0; i < 8; i++) { e[i] = __expf(l[i] - mx); esum += e[i]; }
    const float inv = 1.0f / esum;
    int i1 = 0;
#pragma unroll
    for (int i = 1; i < 8; i++) if (l[i] > l[i1]) i1 = i;
    int i2 = -1;
#pragma unroll
    for (int i = 0; i < 8; i++)
      if (i != i1 && (i2 < 0 || l[i] > l[i2])) i2 = i;
    const float smn  = e[n] * inv;
    const float msum = (e[i1] + e[i2]) * inv;
    const float w    = (n == i1 || n == i2) ? smn / (msum + 1e-8f) : 0.0f;

    float ss = kb * kb;
#pragma unroll
    for (int off = 16; off > 0; off >>= 1) ss += __shfl_xor(ss, off);
    const float kn = kb / (sqrtf(ss) + 1e-6f);

    float retr = 0.0f;
#pragma unroll
    for (int j = 0; j < 32; j++) retr = fmaf(S[j], __shfl(kn, j), retr);
    const float delta = vb - retr;

    if (w > 0.0f) {   // wave-uniform; exact (w==0 => S unchanged)
#pragma unroll
      for (int j = 0; j < 32; j++) {
        const float su = tanhf(fmaf(be, S[j], delta * __shfl(kn, j)));
        S[j] = fmaf(w, su - S[j], S[j]);
      }
    }

    float Sq = 0.0f;
#pragma unroll
    for (int j = 0; j < 32; j++) Sq = fmaf(S[j], __shfl(qv, j), Sq);
    const float sig = 1.0f / (1.0f + __expf(-Sq));
    const float bo  = Sq * (Sq * sig);

    lds_bo[n * 32 + lane] = smn * bo;
    __syncthreads();
    if (n == 0) {
      float o = 0.0f;
#pragma unroll
      for (int nn = 0; nn < 8; nn++) o += lds_bo[nn * 32 + lane];
      out[row * 32 + lane] = o;
    }
    __syncthreads();
  }

  float* sf = out + (size_t)TT * BB * 32 +
              (((size_t)b * NBLOCKS + n) * NSTATE + lane) * NSTATE;
#pragma unroll
  for (int j = 0; j < 32; j++) sf[j] = S[j];
}

extern "C" void kernel_launch(void* const* d_in, const int* in_sizes, int n_in,
                              void* d_out, int out_size, void* d_ws, size_t ws_size,
                              hipStream_t stream) {
  const float* x   = (const float*)d_in[0];
  const float* S0  = (const float*)d_in[1];
  const float* Wr  = (const float*)d_in[2];
  const float* Wkv = (const float*)d_in[3];
  const float* Wb  = (const float*)d_in[4];
  const float* bb  = (const float*)d_in[5];
  const float* Wq  = (const float*)d_in[6];
  float* out = (float*)d_out;

  // workspace carve: logits | kv | beta | q | bf16-swizzled W (~204 MB)
  float* ws      = (float*)d_ws;
  float* logitsb = ws;
  float* kvb     = logitsb + (size_t)NROWS * 8;
  float* betab   = kvb     + (size_t)NROWS * 512;
  float* qb      = betab   + (size_t)NROWS * 256;
  __bf16* bw     = (__bf16*)(qb + (size_t)NROWS * 32);   // NSEG*32*32*16 bf16

  wswizzle_kernel<<<dim3((NSEG * 32 * 32) / 256), dim3(256), 0, stream>>>(
      Wr, Wkv, Wb, Wq, bw);

  proj_wmma_kernel<<<dim3(NROWS / 64), dim3(256), 0, stream>>>(
      x, bw, bb, logitsb, kvb, betab, qb);

  scan_kernel<<<dim3(BB), dim3(256), 0, stream>>>(
      S0, logitsb, kvb, betab, qb, out);
}
